// Mamba1LikeMixer_30923764531648
// MI455X (gfx1250) — matrix operand order repaired and checked by reading, not compiled, not run on hardware
//
#include <hip/hip_runtime.h>
#include <hip/hip_bf16.h>
#include <math.h>

// ---------------------------------------------------------------------------
// Mamba1-like mixer for MI455X (gfx1250, wave32, WMMA).
// GEMMs: v_wmma_f32_16x16x32_bf16, double-buffered LDS, A tiles staged with
// CDNA5 async global->LDS DMA (ASYNCcnt), B tiles converted f32->bf16 with
// packed stores, one barrier per K-step.
// ---------------------------------------------------------------------------

typedef __bf16 bf16_t;
typedef __attribute__((ext_vector_type(16))) __bf16 v16bf;
typedef __attribute__((ext_vector_type(8)))  float  v8f;

#define BM 128
#define BN 128
#define BK 32
#define LDSP 40   // LDS row pitch in bf16 (80B: 16B-aligned, bank-spread)

union FragBF { uint4 u4[2]; v16bf v; };

// CDNA5 async copy: 16B global -> LDS, tracked with ASYNCcnt (no VGPR data).
__device__ __forceinline__ void async_copy_b128(unsigned lds_addr, const void* gaddr)
{
    asm volatile("global_load_async_to_lds_b128 %0, %1, off"
                 :: "v"(lds_addr), "v"((unsigned long long)(uintptr_t)gaddr)
                 : "memory");
}
__device__ __forceinline__ void wait_asynccnt0()
{
    asm volatile("s_wait_asynccnt 0x0" ::: "memory");
}
// Flat LDS pointer -> raw LDS byte address (low 32 bits of the flat address).
__device__ __forceinline__ unsigned lds_addr_of(const void* p)
{
    return (unsigned)(uintptr_t)p;
}

// C[M,N] = A(bf16,[M,K]) @ B(f32,[K,N]) (+bias, +epilogue)
// MODE 0: store bf16            MODE 1: +bias, store bf16
// MODE 2: +bias, softplus+1e-4, store f32
// MODE 3: +bias, store f32
template<int MODE>
__global__ __launch_bounds__(256)
void gemm_bf16_wmma(const bf16_t* __restrict__ A,
                    const float*  __restrict__ B,
                    const float*  __restrict__ bias,
                    void* __restrict__ Cout,
                    int M, int N, int K, int ldc)
{
    __shared__ __align__(16) bf16_t Alds[2][BM * LDSP];   // [m][k]
    __shared__ __align__(16) bf16_t Blds[2][BN * LDSP];   // [n][k] (transposed)

    const int tid  = threadIdx.x;
    const int lane = tid & 31;
    const int wave = tid >> 5;        // 8 waves
    const int wm   = wave >> 1;       // 0..3  -> 32-row strip
    const int wn   = wave & 1;        // 0..1  -> 64-col strip

    const int m0 = blockIdx.x * BM;   // M-tile fastest: B N-slab reused in L2
    const int n0 = blockIdx.y * BN;

    v8f acc[2][4];
#pragma unroll
    for (int i = 0; i < 2; ++i)
#pragma unroll
        for (int j = 0; j < 4; ++j)
            acc[i][j] = (v8f){0.f,0.f,0.f,0.f,0.f,0.f,0.f,0.f};

    // A staging: 2 threads/row, 32B each -> two async b128 DMAs
    const int a_r = tid >> 1;                 // 0..127
    const int a_c = (tid & 1) * 16;           // bf16 col 0 or 16
    // B staging: thread owns k-pair (2*b_kp, 2*b_kp+1) x 8 columns
    const int b_kp = tid >> 4;                // 0..15
    const int b_nn = (tid & 15) * 8;          // 0..120

    auto stageA_async = [&](int k0, int bb) {
        const bf16_t* g = A + (size_t)(m0 + a_r) * K + k0 + a_c;
        bf16_t* l = &Alds[bb][a_r * LDSP + a_c];
        async_copy_b128(lds_addr_of(l),     g);
        async_copy_b128(lds_addr_of(l + 8), g + 8);
    };
    auto stageB_load = [&](int k0, float4* r) {
        const float* row0 = B + (size_t)(k0 + 2 * b_kp) * N + n0 + b_nn;
        r[0] = ((const float4*)row0)[0];
        r[1] = ((const float4*)row0)[1];
        r[2] = ((const float4*)(row0 + N))[0];
        r[3] = ((const float4*)(row0 + N))[1];
    };
    auto stageB_store = [&](int bb, const float4* r) {
        float f0[8], f1[8];
        *(float4*)&f0[0] = r[0]; *(float4*)&f0[4] = r[1];
        *(float4*)&f1[0] = r[2]; *(float4*)&f1[4] = r[3];
#pragma unroll
        for (int j = 0; j < 8; ++j) {
            union { bf16_t h[2]; unsigned u; } p;
            p.h[0] = (bf16_t)f0[j];       // k   = 2*b_kp
            p.h[1] = (bf16_t)f1[j];       // k+1 = 2*b_kp+1
            *(unsigned*)&Blds[bb][(b_nn + j) * LDSP + 2 * b_kp] = p.u;
        }
    };

    // ---- prologue: stage tile 0 into buffer 0 ----
    int buf = 0;
    {
        float4 r[4];
        stageA_async(0, 0);
        stageB_load(0, r);
        stageB_store(0, r);
        wait_asynccnt0();
        __syncthreads();
    }

    for (int k0 = 0; k0 < K; k0 += BK) {
        const bool hn   = (k0 + BK) < K;
        const int  nbuf = buf ^ 1;
        float4 r[4];
        if (hn) {
            stageA_async(k0 + BK, nbuf);   // DMA next A tile (overlaps WMMA)
            stageB_load(k0 + BK, r);       // prefetch next B tile into regs
        }

        // ---- fragments (ISA 16-bit A 16x32 / B 32x16 lane layouts) ----
        // A lane l: row = (l&15), K-runs {kb..kb+7},{kb+16..kb+23}, kb=(l>=16)*8
        // B lane l: col = (l&15), K-runs {kb..kb+7},{kb+16..kb+23}, kb=(l>=16)*8
        FragBF afrag[2], bfrag[4];
        const int ar  = wm * 32 + (lane & 15);
        const int akb = (lane >> 4) * 8;
#pragma unroll
        for (int mi = 0; mi < 2; ++mi) {
            const bf16_t* p = &Alds[buf][(ar + mi * 16) * LDSP + akb];
            afrag[mi].u4[0] = *(const uint4*)(p);
            afrag[mi].u4[1] = *(const uint4*)(p + 16);
        }
        const int bkh = (lane >> 4) * 8;
#pragma unroll
        for (int ni = 0; ni < 4; ++ni) {
            const bf16_t* p = &Blds[buf][(wn * 64 + ni * 16 + (lane & 15)) * LDSP + bkh];
            bfrag[ni].u4[0] = *(const uint4*)(p);
            bfrag[ni].u4[1] = *(const uint4*)(p + 16);
        }

#pragma unroll
        for (int mi = 0; mi < 2; ++mi)
#pragma unroll
            for (int ni = 0; ni < 4; ++ni)
                acc[mi][ni] = __builtin_amdgcn_wmma_f32_16x16x32_bf16(
                    false, afrag[mi].v, false, bfrag[ni].v,
                    (short)0, acc[mi][ni], false, false);

        if (hn) {
            stageB_store(nbuf, r);         // convert+pack into back buffer
            wait_asynccnt0();              // own A DMAs complete
        }
        __syncthreads();                   // publish back buffer
        buf = nbuf;
    }

    // ---- epilogue: D layout VGPR r -> M = r + 8*(lane>=16), N = lane&15 ----
    const int eroff = (lane >> 4) << 3;
    const int ecol  = lane & 15;
#pragma unroll
    for (int mi = 0; mi < 2; ++mi) {
        const int rbase = m0 + wm * 32 + mi * 16 + eroff;
#pragma unroll
        for (int ni = 0; ni < 4; ++ni) {
            const int col = n0 + wn * 64 + ni * 16 + ecol;
            const float bv = (MODE == 0) ? 0.f : bias[col];
#pragma unroll
            for (int r = 0; r < 8; ++r) {
                const float v = acc[mi][ni][r];
                const size_t off = (size_t)(rbase + r) * ldc + col;
                if (MODE == 0) {
                    ((bf16_t*)Cout)[off] = (bf16_t)v;
                } else if (MODE == 1) {
                    ((bf16_t*)Cout)[off] = (bf16_t)(v + bv);
                } else if (MODE == 2) {
                    const float t  = v + bv;
                    const float sp = (t > 20.f) ? t : log1pf(__expf(t));
                    ((float*)Cout)[off] = sp + 1e-4f;
                } else {
                    ((float*)Cout)[off] = v + bv;
                }
            }
        }
    }
}

// fp32 -> bf16 bulk convert (for input x)
__global__ __launch_bounds__(256)
void cvt_f32_bf16(const float* __restrict__ in, bf16_t* __restrict__ out, int n)
{
    const int i = blockIdx.x * blockDim.x + threadIdx.x;
    if (i < n) out[i] = (bf16_t)in[i];
}

// depthwise causal conv (k=4) + bias + SiLU; x_br = cols [0,2048) of xz[L,4096]
__global__ __launch_bounds__(256)
void conv_silu_kernel(const bf16_t* __restrict__ xz,
                      const float* __restrict__ cw,   // [2048][1][4]
                      const float* __restrict__ cb,   // [2048]
                      bf16_t* __restrict__ v)         // [L,2048]
{
    const int d = blockIdx.x * blockDim.x + threadIdx.x;   // 0..2047
    const int l = blockIdx.y;                              // 0..2047
    float acc = cb[d];
#pragma unroll
    for (int j = 0; j < 4; ++j) {
        const int t = l - 3 + j;
        if (t >= 0)
            acc += (float)xz[(size_t)t * 4096 + d] * cw[d * 4 + j];
    }
    const float s = acc / (1.f + __expf(-acc));
    v[(size_t)l * 2048 + d] = (bf16_t)s;
}

// selective scan: lane handles one (d,s); 16 lanes/channel; shfl_xor reduce;
// fused D*u skip, sigmoid(z) gate, bf16 output.
__global__ __launch_bounds__(256)
void scan_kernel(const float*  __restrict__ dt,    // [L,2048] f32 (softplus'd)
                 const bf16_t* __restrict__ u,     // [L,2048]
                 const bf16_t* __restrict__ Bt,    // [L,2048,16]
                 const bf16_t* __restrict__ Ct,    // [L,2048,16]
                 const bf16_t* __restrict__ xz,    // z at cols [2048,4096) of [L,4096]
                 const float*  __restrict__ A_log, // [2048,16]
                 const float*  __restrict__ Dp,    // [2048]
                 bf16_t* __restrict__ y)           // [L,2048]
{
    const int s = threadIdx.x & 15;
    const int d = blockIdx.x * 16 + (threadIdx.x >> 4);
    const float Ac = -__expf(A_log[d * 16 + s]);
    const float Dd = Dp[d];
    float state = 0.f;
    for (int l = 0; l < 2048; ++l) {
        const size_t base = (size_t)l * 2048 + d;
        const float dtv = dt[base];
        const float uv  = (float)u[base];
        const float b   = (float)Bt[base * 16 + s];
        const float c   = (float)Ct[base * 16 + s];
        state = __expf(dtv * Ac) * state + b * uv;
        float p = state * c;
        p += __shfl_xor(p, 8, 16);
        p += __shfl_xor(p, 4, 16);
        p += __shfl_xor(p, 2, 16);
        p += __shfl_xor(p, 1, 16);
        if (s == 0) {
            const float zg = (float)xz[(size_t)l * 4096 + 2048 + d];
            const float yv = (p + Dd * uv) / (1.f + __expf(-zg));
            y[base] = (bf16_t)yv;
        }
    }
}

extern "C" void kernel_launch(void* const* d_in, const int* in_sizes, int n_in,
                              void* d_out, int out_size, void* d_ws, size_t ws_size,
                              hipStream_t stream)
{
    const float* x      = (const float*)d_in[0];
    const float* in_w   = (const float*)d_in[1];
    const float* in_b   = (const float*)d_in[2];
    const float* conv_w = (const float*)d_in[3];
    const float* conv_b = (const float*)d_in[4];
    const float* xp_w   = (const float*)d_in[5];
    const float* dt_w   = (const float*)d_in[6];
    const float* dt_b   = (const float*)d_in[7];
    const float* b_w    = (const float*)d_in[8];
    const float* c_w    = (const float*)d_in[9];
    const float* A_log  = (const float*)d_in[10];
    const float* Dp     = (const float*)d_in[11];
    const float* out_w  = (const float*)d_in[12];
    const float* out_b  = (const float*)d_in[13];

    const int L = 2048, DM = 1024, DI = 2048, DS = 16;

    char*  ws  = (char*)d_ws;
    size_t off = 0;
    auto alloc = [&](size_t bytes) -> char* {
        char* p = ws + off;
        off += (bytes + 255) & ~(size_t)255;
        return p;
    };
    bf16_t* xbf  = (bf16_t*)alloc((size_t)L * DM * 2);        //   4 MB
    bf16_t* xzbf = (bf16_t*)alloc((size_t)L * 2 * DI * 2);    //  16 MB
    bf16_t* vbf  = (bf16_t*)alloc((size_t)L * DI * 2);        //   8 MB
    bf16_t* ubf  = (bf16_t*)alloc((size_t)L * DI * 2);        //   8 MB
    float*  dtf  = (float*)alloc((size_t)L * DI * 4);         //  16 MB
    bf16_t* Btb  = (bf16_t*)alloc((size_t)L * DI * DS * 2);   // 128 MB
    bf16_t* Ctb  = (bf16_t*)alloc((size_t)L * DI * DS * 2);   // 128 MB
    bf16_t* ybf  = (bf16_t*)alloc((size_t)L * DI * 2);        //   8 MB
    (void)ws_size; (void)in_sizes; (void)n_in; (void)out_size;

    // 1) x -> bf16
    {
        const int n = L * DM;
        cvt_f32_bf16<<<(n + 255) / 256, 256, 0, stream>>>(x, xbf, n);
    }
    // 2) xz = x @ in_w + in_b            [2048,4096] bf16
    gemm_bf16_wmma<1><<<dim3(L / BM, (2 * DI) / BN), 256, 0, stream>>>(
        xbf, in_w, in_b, xzbf, L, 2 * DI, DM, 2 * DI);
    // 3) v = silu(causal_conv(x_br) + conv_b)
    conv_silu_kernel<<<dim3(DI / 256, L), 256, 0, stream>>>(xzbf, conv_w, conv_b, vbf);
    // 4) u = v @ xp_w                    [2048,2048] bf16
    gemm_bf16_wmma<0><<<dim3(L / BM, DI / BN), 256, 0, stream>>>(
        vbf, xp_w, nullptr, ubf, L, DI, DI, DI);
    // 5) dt = softplus(u @ dt_w + dt_b) + 1e-4   f32
    gemm_bf16_wmma<2><<<dim3(L / BM, DI / BN), 256, 0, stream>>>(
        ubf, dt_w, dt_b, dtf, L, DI, DI, DI);
    // 6) Bt = u @ b_w                    [2048,32768] bf16
    gemm_bf16_wmma<0><<<dim3(L / BM, (DI * DS) / BN), 256, 0, stream>>>(
        ubf, b_w, nullptr, Btb, L, DI * DS, DI, DI * DS);
    // 7) Ct = u @ c_w
    gemm_bf16_wmma<0><<<dim3(L / BM, (DI * DS) / BN), 256, 0, stream>>>(
        ubf, c_w, nullptr, Ctb, L, DI * DS, DI, DI * DS);
    // 8) selective scan + gate -> y bf16
    scan_kernel<<<DI / 16, 256, 0, stream>>>(dtf, ubf, Btb, Ctb, xzbf, A_log, Dp, ybf);
    // 9) out = y @ out_w + out_b         f32 -> d_out
    gemm_bf16_wmma<3><<<dim3(L / BM, DM / BN), 256, 0, stream>>>(
        ybf, out_w, out_b, (float*)d_out, L, DM, DI, DM);
}
